// LlamaAttention_7352984010786
// MI455X (gfx1250) — compile-verified, matrix operand order checked by reading, not run on hardware
//
#include <hip/hip_runtime.h>
#include <hip/hip_bf16.h>

// ---------------------------------------------------------------------------
// LlamaAttention (with RoPE-overwritten q/k) for MI455X / gfx1250.
//
// Math: scores[i,j] = sum_d cos((i-j)*f_d)  -> Toeplitz, g(0)=384 is the row
// max of every row. softmax row i = E[i-j+8191] / Z_i with E = exp(g-384) and
// Z_i a windowed prefix sum of E. Heavy compute (P@V + projections, ~82 GFLOP
// vs ~120MB traffic -> compute bound) runs on v_wmma_f32_16x16x32_f16.
// ---------------------------------------------------------------------------

typedef __attribute__((ext_vector_type(16))) _Float16 v16h;
typedef __attribute__((ext_vector_type(8)))  _Float16 v8h;
typedef __attribute__((ext_vector_type(8)))  float    v8f;

#define HIDDEN 768
#define KV     256
#define BATCH  2
#define SEQ    8192
#define MTOT   (BATCH * SEQ)
#define ETAB   16384          // 16383 used, last entry padded to 0

static __device__ __forceinline__ v16h cat8(v8h lo, v8h hi) {
  return __builtin_shufflevector(lo, hi, 0,1,2,3,4,5,6,7,8,9,10,11,12,13,14,15);
}

// ------------------------------ stage 0: f32 -> f16 ------------------------
__global__ void cvt_kernel(const float* __restrict__ in, _Float16* __restrict__ out, int n) {
  int i = blockIdx.x * blockDim.x + threadIdx.x;
  int stride = gridDim.x * blockDim.x;
  for (; i < n; i += stride) out[i] = (_Float16)in[i];
}

// ------------------------------ stage 1: Toeplitz table --------------------
// E[d] = exp(g(d-8191) - 384),  g(delta) = sum_{t=0}^{383} cos(delta * 10000^{-t/384})
__global__ void rope_e_kernel(float* __restrict__ Ef32, _Float16* __restrict__ Ef16) {
  int d = blockIdx.x * 256 + threadIdx.x;
  if (d >= ETAB) return;
  if (d == ETAB - 1) { Ef32[d] = 0.0f; Ef16[d] = (_Float16)0.0f; return; } // pad
  float delta = (float)(d - (SEQ - 1));
  const float c = 9.210340372f / 384.0f;   // ln(10000)/384
  float g = 0.0f;
  #pragma unroll 4
  for (int t = 0; t < 384; ++t) {
    float freq = __expf(-c * (float)t);    // 10000^{-t/384}
    g += __cosf(delta * freq);
  }
  float e = __expf(g - 384.0f);            // row max is exactly g(0)=384
  Ef32[d] = e;
  Ef16[d] = (_Float16)e;
}

// ------------------------------ stage 2: scan -> 1/Z_i ---------------------
__global__ __launch_bounds__(1024) void scan_z_kernel(const float* __restrict__ E,
                                                      float* __restrict__ C,
                                                      float* __restrict__ Zinv) {
  __shared__ float part[1024];
  const int t = threadIdx.x;
  float local[16];
  float s = 0.0f;
  #pragma unroll
  for (int k = 0; k < 16; ++k) { local[k] = E[t * 16 + k]; s += local[k]; }
  part[t] = s;
  __syncthreads();
  for (int off = 1; off < 1024; off <<= 1) {     // inclusive scan of partials
    float v = (t >= off) ? part[t - off] : 0.0f;
    __syncthreads();
    part[t] += v;
    __syncthreads();
  }
  float run = (t == 0) ? 0.0f : part[t - 1];
  #pragma unroll
  for (int k = 0; k < 16; ++k) { run += local[k]; C[t * 16 + k] = run; }
  __syncthreads();
  for (int i = t; i < SEQ; i += 1024) {          // Z_i = sum_{j} E[i-j+8191]
    float z = C[i + SEQ - 1] - (i ? C[i - 1] : 0.0f);
    Zinv[i] = 1.0f / z;
  }
}

// ------------------------------ stage 3: V = x @ Wv^T (stored transposed) --
// Vt[b][n][s] = sum_k x[b][s][k] * Wv[n][k]
__global__ __launch_bounds__(256) void vproj_kernel(const _Float16* __restrict__ X,
                                                    const _Float16* __restrict__ Wv,
                                                    _Float16* __restrict__ Vt) {
  const int lane = threadIdx.x & 31;
  const int wave = threadIdx.x >> 5;
  const int r    = lane & 15;
  const int part = lane >> 4;
  const int tile = blockIdx.x * 8 + wave;
  const int TN   = KV / 16;
  const int m0   = (tile / TN) * 16;
  const int n0   = (tile % TN) * 16;

  const _Float16* arow = X  + (size_t)(m0 + r) * HIDDEN;
  const _Float16* brow = Wv + (size_t)(n0 + r) * HIDDEN;
  v8f acc = {};
  for (int k0 = 0; k0 < HIDDEN; k0 += 32) {
    v8h alo = *(const v8h*)(arow + k0 + 8 * part);
    v8h ahi = *(const v8h*)(arow + k0 + 16 + 8 * part);
    v8h blo = *(const v8h*)(brow + k0 + 8 * part);
    v8h bhi = *(const v8h*)(brow + k0 + 16 + 8 * part);
    acc = __builtin_amdgcn_wmma_f32_16x16x32_f16(false, cat8(alo, ahi),
                                                 false, cat8(blo, bhi),
                                                 (short)0, acc, false, false);
  }
  #pragma unroll
  for (int v = 0; v < 8; ++v) {
    int m = m0 + v + 8 * part;
    int b = m >> 13, s = m & (SEQ - 1);
    Vt[((size_t)b * KV + n0 + r) * SEQ + s] = (_Float16)acc[v];
  }
}

// ------------------------------ stage 4: attn = softmax(Toeplitz) @ V ------
// Attn[b][i][n] = Zinv[i] * sum_j E[i-j+8191] * V[b][j][n]
// 512 threads = 16 waves (one per n-tile); each wave does 4 i-tiles (64 rows)
// to amortize the streaming V operand. A-operand is gathered from the 32KB
// E table resident in LDS (Toeplitz diagonal indexing).
__global__ __launch_bounds__(512) void attn_kernel(const _Float16* __restrict__ Ef16,
                                                   const _Float16* __restrict__ Vt,
                                                   const float* __restrict__ Zinv,
                                                   _Float16* __restrict__ Attn) {
  __shared__ _Float16 elds[ETAB];
  const int tid = threadIdx.x;
  for (int t = tid; t < ETAB; t += 512) elds[t] = Ef16[t];
  __syncthreads();

  const int lane = tid & 31;
  const int wave = tid >> 5;
  const int r    = lane & 15;
  const int part = lane >> 4;
  const int b    = blockIdx.x >> 7;            // 128 row-blocks per batch
  const int i0   = (blockIdx.x & 127) * 64;
  const int n0   = wave * 16;

  const _Float16* vrow = Vt + ((size_t)b * KV + n0 + r) * SEQ;
  v8f acc[4] = {};
  const int ibase = (SEQ - 1) + i0 + r;

  for (int j0 = 0; j0 < SEQ; j0 += 32) {
    __builtin_prefetch(vrow + j0 + 512, 0, 0);          // global_prefetch_b8
    v8h blo = *(const v8h*)(vrow + j0 + 8 * part);
    v8h bhi = *(const v8h*)(vrow + j0 + 16 + 8 * part);
    v16h bm = cat8(blo, bhi);
    #pragma unroll
    for (int t = 0; t < 4; ++t) {
      const _Float16* ap = elds + (ibase + t * 16 - j0 - 8 * part);
      v16h am;
      #pragma unroll
      for (int jj = 0; jj < 8; ++jj) {
        am[jj]     = ap[-jj];          // k = 8*part + jj
        am[8 + jj] = ap[-16 - jj];     // k = 16 + 8*part + jj
      }
      acc[t] = __builtin_amdgcn_wmma_f32_16x16x32_f16(false, am, false, bm,
                                                      (short)0, acc[t], false, false);
    }
  }
  #pragma unroll
  for (int t = 0; t < 4; ++t) {
    #pragma unroll
    for (int v = 0; v < 8; ++v) {
      int i = i0 + t * 16 + v + 8 * part;
      float val = acc[t][v] * Zinv[i];
      Attn[((size_t)b * SEQ + i) * KV + n0 + r] = (_Float16)val;
    }
  }
}

// ------------------------------ stage 5: out = attn @ Wo^T -----------------
__global__ __launch_bounds__(256) void oproj_kernel(const _Float16* __restrict__ A,
                                                    const _Float16* __restrict__ Wo,
                                                    float* __restrict__ Out) {
  const int lane = threadIdx.x & 31;
  const int wave = threadIdx.x >> 5;
  const int r    = lane & 15;
  const int part = lane >> 4;
  const int tile = blockIdx.x * 8 + wave;
  const int TN   = HIDDEN / 16;                 // 48
  const int m0   = (tile / TN) * 16;
  const int n0   = (tile % TN) * 16;

  const _Float16* arow = A  + (size_t)(m0 + r) * KV;
  const _Float16* brow = Wo + (size_t)(n0 + r) * KV;
  v8f acc = {};
  #pragma unroll
  for (int k0 = 0; k0 < KV; k0 += 32) {
    v8h alo = *(const v8h*)(arow + k0 + 8 * part);
    v8h ahi = *(const v8h*)(arow + k0 + 16 + 8 * part);
    v8h blo = *(const v8h*)(brow + k0 + 8 * part);
    v8h bhi = *(const v8h*)(brow + k0 + 16 + 8 * part);
    acc = __builtin_amdgcn_wmma_f32_16x16x32_f16(false, cat8(alo, ahi),
                                                 false, cat8(blo, bhi),
                                                 (short)0, acc, false, false);
  }
  #pragma unroll
  for (int v = 0; v < 8; ++v) {
    int m = m0 + v + 8 * part;
    Out[(size_t)m * HIDDEN + n0 + r] = acc[v];
  }
}

// ---------------------------------------------------------------------------
extern "C" void kernel_launch(void* const* d_in, const int* in_sizes, int n_in,
                              void* d_out, int out_size, void* d_ws, size_t ws_size,
                              hipStream_t stream) {
  (void)in_sizes; (void)n_in; (void)out_size; (void)ws_size;
  const float* x   = (const float*)d_in[0];   // [2,8192,768]
  const float* Wv  = (const float*)d_in[3];   // [256,768]  (Wq/Wk are dead code)
  const float* Wo  = (const float*)d_in[4];   // [768,256]

  char* w = (char*)d_ws;
  size_t off = 0;
  auto alloc = [&](size_t bytes) -> void* {
    void* p = w + off;
    off = (off + bytes + 255) & ~(size_t)255;
    return p;
  };
  _Float16* xf16  = (_Float16*)alloc((size_t)MTOT * HIDDEN * 2);   // 25.2 MB
  _Float16* Wv16  = (_Float16*)alloc((size_t)KV * HIDDEN * 2);
  _Float16* Wo16  = (_Float16*)alloc((size_t)HIDDEN * KV * 2);
  float*    Ef32  = (float*)   alloc((size_t)ETAB * 4);
  _Float16* Ef16  = (_Float16*)alloc((size_t)ETAB * 2);
  float*    Csum  = (float*)   alloc((size_t)ETAB * 4);
  float*    Zinv  = (float*)   alloc((size_t)SEQ * 4);
  _Float16* Vt    = (_Float16*)alloc((size_t)BATCH * KV * SEQ * 2); // 8.4 MB
  _Float16* At16  = (_Float16*)alloc((size_t)BATCH * SEQ * KV * 2); // 8.4 MB

  cvt_kernel<<<2048, 256, 0, stream>>>(x,  xf16, MTOT * HIDDEN);
  cvt_kernel<<<192,  256, 0, stream>>>(Wv, Wv16, KV * HIDDEN);
  cvt_kernel<<<192,  256, 0, stream>>>(Wo, Wo16, HIDDEN * KV);
  rope_e_kernel<<<ETAB / 256, 256, 0, stream>>>(Ef32, Ef16);
  scan_z_kernel<<<1, 1024, 0, stream>>>(Ef32, Csum, Zinv);
  vproj_kernel<<<(MTOT / 16) * (KV / 16) / 8, 256, 0, stream>>>(xf16, Wv16, Vt);
  attn_kernel<<<BATCH * (SEQ / 64), 512, 0, stream>>>(Ef16, Vt, Zinv, At16);
  oproj_kernel<<<(MTOT / 16) * (HIDDEN / 16) / 8, 256, 0, stream>>>(At16, Wo16, (float*)d_out);
}
